// MinimalPhysicsModel_76957224009994
// MI455X (gfx1250) — matrix operand order repaired and checked by reading, not compiled, not run on hardware
//
#include <hip/hip_runtime.h>
#include <math.h>

#define BATCH   131072
#define STEPS   50
#define HID     64
#define DT      (1.0f / 30.0f)
#define EPSF    1e-6f

#define WAVES_PER_BLOCK 4
#define ROWS_PER_WAVE   32            // one row per lane
#define ROWS_PER_BLOCK  (WAVES_PER_BLOCK * ROWS_PER_WAVE)
#define FSTRIDE         12            // feats row stride (floats)
#define HSTRIDE         68            // hidden row stride (floats)

typedef float    v8f  __attribute__((ext_vector_type(8)));
typedef float    v4f  __attribute__((ext_vector_type(4)));
typedef float    v2f  __attribute__((ext_vector_type(2)));
typedef _Float16 v16h __attribute__((ext_vector_type(16)));
typedef __fp16   h2t  __attribute__((ext_vector_type(2)));   // cvt_pkrtz result type

// Compiler-only memory fence (0 instructions). All LDS staging is per-wave and
// per-wave DS ops are in-order in hardware; we only stop compiler reordering.
#define WAVE_FENCE() asm volatile("" ::: "memory")

// tanh(x) = 1 - 2/(exp2(2*log2e*x)+1); v_exp_f32 + v_rcp_f32, saturates correctly.
__device__ __forceinline__ float fast_tanhf(float x) {
    const float e = __builtin_amdgcn_exp2f(x * 2.8853900817779268f);
    const float r = __builtin_amdgcn_rcpf(e + 1.0f);
    return __builtin_fmaf(-2.0f, r, 1.0f);
}

// Polynomial atan2, ~1e-6 rad; inputs here are strictly first-quadrant.
__device__ __forceinline__ float fast_atan2f(float y, float x) {
    const float ax = fabsf(x), ay = fabsf(y);
    const float mx = fmaxf(ax, ay), mn = fminf(ax, ay);
    const float t  = mn * __builtin_amdgcn_rcpf(mx);
    const float s  = t * t;
    float p = __builtin_fmaf(s, -0.01172120f, 0.05265332f);
    p = __builtin_fmaf(s, p, -0.11643287f);
    p = __builtin_fmaf(s, p,  0.19354346f);
    p = __builtin_fmaf(s, p, -0.33262347f);
    p = __builtin_fmaf(s, p,  0.99997726f);
    float r = t * p;
    r = (ay > ax)  ? (1.5707963267948966f - r) : r;
    r = (x < 0.0f) ? (3.1415926535897931f - r) : r;
    r = (y < 0.0f) ? -r : r;
    return r;
}

__device__ __forceinline__ v8f splat8(float v) {
    v8f c;
#pragma unroll
    for (int i = 0; i < 8; ++i) c[i] = v;
    return c;
}

// Build a 16-bit A fragment (16x16x32 layout) from f32 LDS rows:
// hr points at (row base + chunk base + lane half-offset); lane carries
// K = base..base+7 (VGPR0-3) and base+16..base+23 (VGPR4-7), 2 halves/VGPR.
__device__ __forceinline__ v16h load_a_f16(const float* hr) {
    union { v16h v; h2t p[8]; } u;
#pragma unroll
    for (int j = 0; j < 4; ++j) {
        v2f t0 = *(const v2f*)(hr + 2 * j);          // ds_load_b64
        v2f t1 = *(const v2f*)(hr + 16 + 2 * j);     // ds_load_b64
        u.p[j]     = __builtin_amdgcn_cvt_pkrtz(t0.x, t0.y);
        u.p[4 + j] = __builtin_amdgcn_cvt_pkrtz(t1.x, t1.y);
    }
    return u.v;
}

__global__ __launch_bounds__(128)
void physics_mlp_wmma_kernel(const float* __restrict__ gin,
                             const float* __restrict__ W1,
                             const float* __restrict__ b1,
                             const float* __restrict__ W2,
                             const float* __restrict__ b2,
                             const float* __restrict__ W3,
                             const float* __restrict__ b3,
                             const float* __restrict__ grav,
                             const float* __restrict__ fric,
                             float* __restrict__ gout)
{
    __shared__ float sF[WAVES_PER_BLOCK][ROWS_PER_WAVE * FSTRIDE];
    __shared__ float sH[WAVES_PER_BLOCK][ROWS_PER_WAVE * HSTRIDE];
    __shared__ float sC[WAVES_PER_BLOCK][ROWS_PER_WAVE * 2];

    const int  tid  = threadIdx.x;
    const int  lane = tid & 31;
    const int  wave = tid >> 5;
    const int  ln   = lane & 15;
    const bool hi   = lane >= 16;
    const int  khi  = hi ? 2 : 0;

    // Per-lane bias values (column N = nt*16 + ln), folded into WMMA C operand.
    float bias1[4], bias2[4];
#pragma unroll
    for (int nt = 0; nt < 4; ++nt) {
        bias1[nt] = b1[nt * 16 + ln];
        bias2[nt] = b2[nt * 16 + ln];
    }
    const float b3n = (ln == 0) ? b3[0] : ((ln == 1) ? b3[1] : 0.0f);
    const float pixel_gravity = grav[0] * 40.0f;
    const float friction      = fabsf(fric[0]);

    // ---- Layer-1 B-fragments: f32 16x16x4 layout (features need f32 range) ----
    v2f w1f[3][4];                       // K padded 11 -> 12
#pragma unroll
    for (int kc = 0; kc < 3; ++kc) {
#pragma unroll
        for (int nt = 0; nt < 4; ++nt) {
            const int kr  = kc * 4 + khi;
            const int col = nt * 16 + ln;
            w1f[kc][nt].x = (kr     < 11) ? W1[kr * HID + col]       : 0.0f;
            w1f[kc][nt].y = (kr + 1 < 11) ? W1[(kr + 1) * HID + col] : 0.0f;
        }
    }

    // ---- Layer-2/3 B-fragments: f16 16x16x32 layout (tanh-bounded inputs) ----
    // B 32x16: VGPR j = pack(B[k][col], B[k+1][col]), k=base+2j+(j>=4?8:0), base=kc*32+(hi?8:0)
    v16h w2h[2][4];
#pragma unroll
    for (int kc = 0; kc < 2; ++kc) {
#pragma unroll
        for (int nt = 0; nt < 4; ++nt) {
            const int base = kc * 32 + (hi ? 8 : 0);
            const int col  = nt * 16 + ln;
            union { v16h v; h2t p[8]; } u;
#pragma unroll
            for (int j = 0; j < 8; ++j)
                u.p[j] = __builtin_amdgcn_cvt_pkrtz(W2[(base + 2 * j + (j >= 4 ? 8 : 0)) * HID + col],
                                                    W2[(base + 2 * j + (j >= 4 ? 8 : 0) + 1) * HID + col]);
            w2h[kc][nt] = u.v;
        }
    }
    v16h w3h[2];                         // W3 padded 64x2 -> 64x16 (cols>=2 zero)
#pragma unroll
    for (int kc = 0; kc < 2; ++kc) {
        const int base = kc * 32 + (hi ? 8 : 0);
        union { v16h v; h2t p[8]; } u;
#pragma unroll
        for (int j = 0; j < 8; ++j) {
            const int kr = base + 2 * j + (j >= 4 ? 8 : 0);
            const float lo = (ln < 2) ? W3[kr * 2 + ln]       : 0.0f;
            const float hh = (ln < 2) ? W3[(kr + 1) * 2 + ln] : 0.0f;
            u.p[j] = __builtin_amdgcn_cvt_pkrtz(lo, hh);
        }
        w3h[kc] = u.v;
    }

    float* myF = sF[wave];
    float* myH = sH[wave];
    float* myC = sC[wave];

    const int b = blockIdx.x * ROWS_PER_BLOCK + wave * ROWS_PER_WAVE + lane;

    // Initial state = inputs[b, 0, 0:8]; each lane owns one row.
    const float* sp = gin + (size_t)b * (STEPS * 8);
    float sx1, sy1, sx2, sy2, svx1, svy1, svx2, svy2;
    {
        v4f p4 = *(const v4f*)(sp);
        v4f v4 = *(const v4f*)(sp + 4);
        sx1 = p4.x;  sy1 = p4.y;  sx2 = p4.z;  sy2 = p4.w;
        svx1 = v4.x; svy1 = v4.y; svx2 = v4.z; svy2 = v4.w;
    }
    float* op = gout + (size_t)b * (STEPS * 8);

    for (int t = 0; t < STEPS; ++t) {
        // Emit pre-update state (this lane's whole row: 2x b128).
        {
            v4f p4; p4.x = sx1; p4.y = sy1; p4.z = sx2; p4.w = sy2;
            v4f v4; v4.x = svx1; v4.y = svy1; v4.z = svx2; v4.w = svy2;
            *(v4f*)(op + (size_t)t * 8)     = p4;
            *(v4f*)(op + (size_t)t * 8 + 4) = v4;
        }

        const float x1 = sx1, y1 = sy1, x2 = sx2, y2 = sy2;
        const float vx1 = svx1, vy1 = svy1, vx2 = svx2, vy2 = svy2;

        // ---- Physics features (fast sqrt/rcp/atan2; inputs well-conditioned) ----
        const float r1  = __builtin_amdgcn_sqrtf(x1 * x1 + y1 * y1);
        const float th1 = fast_atan2f(y1, x1);
        const float vr1 = (x1 * vx1 + y1 * vy1) * __builtin_amdgcn_rcpf(r1 + EPSF);
        const float vt1 = (x1 * vy1 - y1 * vx1) * __builtin_amdgcn_rcpf(r1 * r1 + EPSF);
        const float L1  = r1 * vt1;
        const float r2  = __builtin_amdgcn_sqrtf(x2 * x2 + y2 * y2);
        const float th2 = fast_atan2f(y2, x2);
        const float vr2 = (x2 * vx2 + y2 * vy2) * __builtin_amdgcn_rcpf(r2 + EPSF);
        const float vt2 = (x2 * vy2 - y2 * vx2) * __builtin_amdgcn_rcpf(r2 * r2 + EPSF);
        const float L2  = r2 * vt2;
        const float dxp = x2 - x1, dyp = y2 - y1;
        const float r12 = __builtin_amdgcn_sqrtf(dxp * dxp + dyp * dyp);

        // Stage feats row-major into LDS (3x ds_store_b128, conflict-free).
        {
            float* fr = myF + lane * FSTRIDE;
            v4f f0; f0.x = r1;  f0.y = th1; f0.z = vr1; f0.w = vt1;
            v4f f1; f1.x = L1;  f1.y = r2;  f1.z = th2; f1.w = vr2;
            v4f f2; f2.x = vt2; f2.y = L2;  f2.z = r12; f2.w = 0.0f;
            *(v4f*)(fr)     = f0;
            *(v4f*)(fr + 4) = f1;
            *(v4f*)(fr + 8) = f2;
        }
        WAVE_FENCE();

        // ---- Layer 1 (f32 WMMA): feats(32x12) @ W1(12x64), two M-tiles ----
#pragma unroll
        for (int mt = 0; mt < 2; ++mt) {
            v8f acc[4];
#pragma unroll
            for (int nt = 0; nt < 4; ++nt) acc[nt] = splat8(bias1[nt]);  // bias as C
#pragma unroll
            for (int kc = 0; kc < 3; ++kc) {
                v2f a = *(const v2f*)(myF + (mt * 16 + ln) * FSTRIDE + kc * 4 + khi);
#pragma unroll
                for (int nt = 0; nt < 4; ++nt)
                    acc[nt] = __builtin_amdgcn_wmma_f32_16x16x4_f32(
                        false, a, false, w1f[kc][nt], (short)0, acc[nt], false, false);
            }
#pragma unroll
            for (int nt = 0; nt < 4; ++nt)
#pragma unroll
                for (int r = 0; r < 8; ++r)
                    myH[(mt * 16 + (hi ? 8 : 0) + r) * HSTRIDE + nt * 16 + ln] =
                        fast_tanhf(acc[nt][r]);
        }
        WAVE_FENCE();

        // ---- Layer 2 (f16 WMMA, f32 accum): h1(32x64) @ W2(64x64) ----
#pragma unroll
        for (int mt = 0; mt < 2; ++mt) {
            v8f acc[4];
#pragma unroll
            for (int nt = 0; nt < 4; ++nt) acc[nt] = splat8(bias2[nt]);
#pragma unroll
            for (int kc = 0; kc < 2; ++kc) {
                v16h a = load_a_f16(myH + (mt * 16 + ln) * HSTRIDE + kc * 32 + (hi ? 8 : 0));
#pragma unroll
                for (int nt = 0; nt < 4; ++nt)
                    acc[nt] = __builtin_amdgcn_wmma_f32_16x16x32_f16(
                        false, a, false, w2h[kc][nt], (short)0, acc[nt], false, false);
            }
            // overwrite h1 tile mt with h2 tile mt (program order; per-wave DS in-order)
#pragma unroll
            for (int nt = 0; nt < 4; ++nt)
#pragma unroll
                for (int r = 0; r < 8; ++r)
                    myH[(mt * 16 + (hi ? 8 : 0) + r) * HSTRIDE + nt * 16 + ln] =
                        fast_tanhf(acc[nt][r]);
        }
        WAVE_FENCE();

        // ---- Layer 3 (f16 WMMA): corr_raw = h2 @ W3pad + b3 (cols 0/1 real) ----
#pragma unroll
        for (int mt = 0; mt < 2; ++mt) {
            v8f acc = splat8(b3n);
#pragma unroll
            for (int kc = 0; kc < 2; ++kc) {
                v16h a = load_a_f16(myH + (mt * 16 + ln) * HSTRIDE + kc * 32 + (hi ? 8 : 0));
                acc = __builtin_amdgcn_wmma_f32_16x16x32_f16(
                    false, a, false, w3h[kc], (short)0, acc, false, false);
            }
            if (ln < 2) {
#pragma unroll
                for (int r = 0; r < 8; ++r)
                    myC[(mt * 16 + (hi ? 8 : 0) + r) * 2 + ln] = acc[r];
            }
        }
        WAVE_FENCE();

        // ---- Integrate + bounce (own row) ----
        v2f cr = *(const v2f*)(myC + lane * 2);
        const float c0 = cr.x * 0.1f;
        const float c1 = cr.y * 0.1f;

        const float a1x = c0 - friction * vx1;
        const float a1y = pixel_gravity - friction * vy1 + c1;
        const float a2x = c0 - friction * vx2;
        const float a2y = pixel_gravity - friction * vy2 + c1;

        float nvx1 = vx1 + a1x * DT, nvy1 = vy1 + a1y * DT;
        float nvx2 = vx2 + a2x * DT, nvy2 = vy2 + a2y * DT;
        float npx1 = x1 + nvx1 * DT, npy1 = y1 + nvy1 * DT;
        float npx2 = x2 + nvx2 * DT, npy2 = y2 + nvy2 * DT;

        nvx1 = (npx1 < 20.0f || npx1 > 780.0f) ? -0.8f * nvx1 : nvx1;
        npx1 = fminf(fmaxf(npx1, 20.0f), 780.0f);
        nvy1 = (npy1 < 20.0f || npy1 > 580.0f) ? -0.8f * nvy1 : nvy1;
        npy1 = fminf(fmaxf(npy1, 20.0f), 580.0f);
        nvx2 = (npx2 < 20.0f || npx2 > 780.0f) ? -0.8f * nvx2 : nvx2;
        npx2 = fminf(fmaxf(npx2, 20.0f), 780.0f);
        nvy2 = (npy2 < 20.0f || npy2 > 580.0f) ? -0.8f * nvy2 : nvy2;
        npy2 = fminf(fmaxf(npy2, 20.0f), 580.0f);

        sx1 = npx1; sy1 = npy1; sx2 = npx2; sy2 = npy2;
        svx1 = nvx1; svy1 = nvy1; svx2 = nvx2; svy2 = nvy2;

        WAVE_FENCE();   // LDS consumers done (program order) before next step
    }
}

extern "C" void kernel_launch(void* const* d_in, const int* in_sizes, int n_in,
                              void* d_out, int out_size, void* d_ws, size_t ws_size,
                              hipStream_t stream) {
    (void)in_sizes; (void)n_in; (void)out_size; (void)d_ws; (void)ws_size;
    const float* gin  = (const float*)d_in[0];
    const float* W1   = (const float*)d_in[1];
    const float* b1   = (const float*)d_in[2];
    const float* W2   = (const float*)d_in[3];
    const float* b2   = (const float*)d_in[4];
    const float* W3   = (const float*)d_in[5];
    const float* b3   = (const float*)d_in[6];
    const float* grav = (const float*)d_in[7];
    const float* fric = (const float*)d_in[8];
    float* gout = (float*)d_out;

    dim3 grid(BATCH / ROWS_PER_BLOCK);
    dim3 block(WAVES_PER_BLOCK * 32);
    physics_mlp_wmma_kernel<<<grid, block, 0, stream>>>(
        gin, W1, b1, W2, b2, W3, b3, grav, fric, gout);
}
